// PaddGRULayer_44470091382994
// MI455X (gfx1250) — matrix operand order repaired and checked by reading, not compiled, not run on hardware
//
#include <hip/hip_runtime.h>

typedef __attribute__((ext_vector_type(16))) _Float16 v16h;
typedef __attribute__((ext_vector_type(8)))  float    v8f;

#define B_  256
#define T_  512
#define F_  64
#define U_  128
#define X5F 320

// packed f16 weight-fragment offsets in d_ws (units: halves)
#define OFS_WREG  0        // 128x64
#define OFS_WCTL  8192     // 128x64
#define OFS_WDEL  16384    // 64x128
#define OFS_WVM   24576    // 128x128
#define OFS_WCORR 40960    // 64x64 (diag masked)
#define OFS_WCAT  45056    // 192x64
#define OFS_UU    57344    // 128x128
#define OFS_WU    73728    // 128x128
#define OFS_UR    90112    // 128x128
#define OFS_WR    106496   // 128x128  (total 122880 halves = 245760 B)

// ---------------------------------------------------------------------------
// Pre-pass: quantize fp32 weights to f16 in WMMA B-fragment order.
// ---------------------------------------------------------------------------
__global__ void pack_weights(const float* __restrict__ W, _Float16* __restrict__ dst,
                             int K, int N, int maskdiag) {
  int e = blockIdx.x * blockDim.x + threadIdx.x;
  if (e >= K * N) return;
  int k = e / N, n = e % N;
  float v = W[e];
  if (maskdiag && k == n) v = 0.f;            // off-diagonal mask for w_corr
  int KC = K >> 5;
  int kc = k >> 5, kin = k & 31;
  int hi = kin >> 4, j = kin & 15;
  int ntile = n >> 4, lanev = (hi << 4) | (n & 15);
  dst[(((ntile * KC + kc) * 32 + lanev) << 4) + j] = (_Float16)v;
}

__device__ __forceinline__ v8f wmma32(v16h a, v16h b, v8f c) {
  return __builtin_amdgcn_wmma_f32_16x16x32_f16(false, a, false, b, (short)0, c, false, false);
}

// element (row m, col k) inside an f16 A-fragment-ordered LDS buffer
__device__ __forceinline__ int apos(int m, int k) {
  int kc = k >> 5, kin = k & 31;
  return (kc << 9) + ((m + (kin & 16)) << 4) + (kin & 15);
}

__device__ __forceinline__ v16h load_af(const _Float16* ah, int kc) {
  int lane = threadIdx.x & 31;
  return *(const v16h*)(ah + (kc << 9) + (lane << 4));
}

__device__ __forceinline__ v16h load_b(const _Float16* __restrict__ ws, int ofs, int KC, int kc, int ntile) {
  int lane = threadIdx.x & 31;
  return *(const v16h*)(ws + ofs + (((ntile * KC + kc) * 32 + lane) << 4));
}

// fast transcendental path: v_exp_f32 + v_rcp_f32 (no div_fixup chains)
__device__ __forceinline__ float fast_sigmoid(float x) {
  return __builtin_amdgcn_rcpf(1.f + __expf(-x));
}
__device__ __forceinline__ float fast_tanh(float x) {
  return 2.f * __builtin_amdgcn_rcpf(1.f + __expf(-2.f * x)) - 1.f;
}

// ---------------------------------------------------------------------------
// Persistent recurrent kernel: one block = 16 batch rows, loops all T steps.
// 16 waves (512 thr). Wave roles:
//   0-3 : x_reg GEMM | decay | dh-elem | z GEMM | rdh-elem | h_hat+update
//   4-7 : tv GEMM    | decay | dh-elem | z GEMM | rdh-elem | h_hat+update
//   8-11: z_corr GEMM| r GEMM | Pvm GEMM
//   12-15: beta GEMM | r GEMM | Pvm GEMM
// ---------------------------------------------------------------------------
__global__ __launch_bounds__(512, 1)
void padd_gru(const float* __restrict__ inputs,
              const float* __restrict__ h0, const float* __restrict__ ts0,
              const float* __restrict__ dt0,
              const float* __restrict__ b_regression, const float* __restrict__ b_controleur,
              const float* __restrict__ b_delta, const float* __restrict__ b_value_mask,
              const float* __restrict__ b_corr, const float* __restrict__ b_concat,
              const float* __restrict__ b_update, const float* __restrict__ b_reset,
              const _Float16* __restrict__ ws, float* __restrict__ out) {
  // fp32 pool (40 KB) with manual aliasing
  __shared__ float fp[10240];
  float* sh_h      = fp;          // 2048: hidden state
  float* sh_decay  = fp + 2048;   // 2048: decay, later r
  float* sh_ts     = fp + 4096;   // 1024
  float* sh_dt     = fp + 5120;   // 1024
  float* sh_xreg   = fp + 6144;   // 1024
  float* sh_tvbeta = fp + 7168;   // 1024: tv, later beta
  float* sh_xcur   = fp + 8192;   // 1024
  float* sh_zcorr  = fp + 9216;   // 1024
  float* sh_pvm    = fp + 6144;   // 2048: Pvm = cm@Wvm+b (aliases xreg+tvbeta)
  // f16 A-operand mirrors in WMMA fragment order (22 KB)
  __shared__ alignas(32) _Float16 ah_h[16 * 128];
  __shared__ alignas(32) _Float16 ah_dt[16 * 64];
  __shared__ alignas(32) _Float16 ah_xcur[16 * 64];
  __shared__ alignas(32) _Float16 ah_m[16 * 64];
  __shared__ alignas(32) _Float16 ah_decay[16 * 128];
  __shared__ alignas(32) _Float16 ah_dh[16 * 128];   // dh, later r*dh (in place)
  __shared__ alignas(32) _Float16 ah_cm[16 * 128];   // [c | m]

  const int tid  = threadIdx.x;
  const int wq   = tid >> 5;
  const int wqu  = __builtin_amdgcn_readfirstlane(wq);   // scalar wave id
  const int lane = tid & 31;
  const int laneN = lane & 15;
  const int rbase = (lane >> 4) << 3;
  const int ntF = wq & 3;
  const int b0  = blockIdx.x << 4;

  const float b_reg0 = b_regression[0], b_del0 = b_delta[0], b_vm0 = b_value_mask[0];
  const float b_cat0 = b_concat[0], b_upd0 = b_update[0], b_rst0 = b_reset[0];
  const int colF = (ntF << 4) + laneN;
  const int colU  = (wq << 4) + laneN;          // waves 0-7
  const int colU2 = ((wq & 7) << 4) + laneN;    // waves 8-15
  const float b_ctl_c = b_controleur[colF];
  const float b_cor_c = b_corr[colF];

  // ---- hoist weight B-fragments into registers (role-specialized; only
  //      the slots each role actually reads are initialized/live) ----
  v16h Bx[6], By[4], Bz[4], Bv[4];
  if (wqu < 4) {
#pragma unroll
    for (int kc = 0; kc < 4; ++kc) Bx[kc] = load_b(ws, OFS_WREG, 4, kc, ntF);
  } else if (wqu < 8) {
#pragma unroll
    for (int kc = 0; kc < 4; ++kc) Bx[kc] = load_b(ws, OFS_WCTL, 4, kc, ntF);
  } else if (wqu < 12) {
    Bx[0] = load_b(ws, OFS_WCORR, 2, 0, ntF);
    Bx[1] = load_b(ws, OFS_WCORR, 2, 1, ntF);
  } else {
#pragma unroll
    for (int i = 0; i < 6; ++i) Bx[i] = load_b(ws, OFS_WCAT, 6, i, ntF);
  }
  if (wqu < 8) {
#pragma unroll
    for (int kc = 0; kc < 4; ++kc) {
      By[kc] = load_b(ws, OFS_UU, 4, kc, wq);
      Bz[kc] = load_b(ws, OFS_WU, 4, kc, wq);
    }
    Bv[0] = load_b(ws, OFS_WDEL, 2, 0, wq);
    Bv[1] = load_b(ws, OFS_WDEL, 2, 1, wq);
  } else {
    int nt = wq & 7;
#pragma unroll
    for (int kc = 0; kc < 4; ++kc) {
      By[kc] = load_b(ws, OFS_UR, 4, kc, nt);
      Bz[kc] = load_b(ws, OFS_WR, 4, kc, nt);
      Bv[kc] = load_b(ws, OFS_WVM, 4, kc, nt);
    }
  }

  // ---- load initial state (f32 + f16 A-mirror for h) ----
  for (int i = tid; i < 16 * 128; i += 512) {
    int row = i >> 7, u = i & 127;
    float hv = h0[(size_t)(b0 + row) * 128 + u];
    sh_h[i] = hv;
    ah_h[apos(row, u)] = (_Float16)hv;
  }
  for (int i = tid; i < 16 * 64; i += 512) {
    size_t g = (size_t)(b0 + (i >> 6)) * 64 + (i & 63);
    sh_ts[i] = ts0[g];
    sh_dt[i] = dt0[g];
  }
  __syncthreads();

  for (int t = 0; t < T_; ++t) {
    // ---- stage 2: x_reg = h@w_reg (w0-3); tv = |h@w_ctl+b| (w4-7)
    if (wqu < 8) {
      v8f acc = {};
#pragma unroll
      for (int kc = 0; kc < 4; ++kc) acc = wmma32(load_af(ah_h, kc), Bx[kc], acc);
      if (wqu < 4) {
#pragma unroll
        for (int rr = 0; rr < 8; ++rr)
          sh_xreg[(rbase + rr) * 64 + colF] = acc[rr] + b_reg0;
      } else {
#pragma unroll
        for (int rr = 0; rr < 8; ++rr)
          sh_tvbeta[(rbase + rr) * 64 + colF] = fabsf(acc[rr] + b_ctl_c);
      }
    }
    __syncthreads();

    // ---- stage 3: gating elementwise; f16 mirrors; defer out stores
    float o_xr[2], o_ti[2], o_dt[2];
#pragma unroll
    for (int i = 0; i < 2; ++i) {
      int idx = tid + i * 512;
      int row = idx >> 6, f = idx & 63;
      const float* xr = inputs + ((size_t)(b0 + row) * T_ + t) * X5F;
      if (t + 1 < T_ && (f & 31) == 0) {
#pragma unroll
        for (int s = 0; s < 5; ++s) __builtin_prefetch(xr + X5F + s * 64 + f, 0, 0);
      }
      float ti_in = xr[f], x_in = xr[64 + f], mm = xr[128 + f];
      float dt_in = xr[192 + f], pad = xr[256 + f];
      float pts = sh_ts[idx], pdt = sh_dt[idx];
      float tvv = sh_tvbeta[idx], xrg = sh_xreg[idx];
      float sum = pts + tvv;
      float ti = pad * ti_in + (1.f - pad) * sum;
      ti = (ti <= 100.f) ? ti : pts;
      float dgen = (sum <= 100.f) ? tvv : pdt;
      float dtn = pad * dt_in + (1.f - pad) * dgen;
      float xc = mm * x_in + (1.f - mm) * xrg;
      sh_ts[idx] = ti; sh_dt[idx] = dtn; sh_xcur[idx] = xc;
      int pf = apos(row, f);
      ah_dt[pf]   = (_Float16)dtn;
      ah_xcur[pf] = (_Float16)xc;
      ah_m[pf]    = (_Float16)mm;
      ah_cm[apos(row, 64 + f)] = (_Float16)mm;   // m-half of [c|m]
      o_xr[i] = xrg; o_ti[i] = ti; o_dt[i] = dtn;
    }
    __syncthreads();
    // deferred stage-3 output stores (overlap with stage-4 GEMMs)
#pragma unroll
    for (int i = 0; i < 2; ++i) {
      int idx = tid + i * 512;
      int row = idx >> 6, f = idx & 63;
      float* ob = out + ((size_t)t * B_ + (b0 + row)) * X5F;
      ob[f] = o_xr[i]; ob[192 + f] = o_ti[i]; ob[256 + f] = o_dt[i];
    }

    // ---- stage 4: decay (w0-7); z_corr (w8-11)
    v8f zc;   // z_corr values carried across barrier for deferred store
    if (wqu < 8) {
      v8f acc;
#pragma unroll
      for (int e = 0; e < 8; ++e) acc[e] = b_del0;
      acc = wmma32(load_af(ah_dt, 0), Bv[0], acc);
      acc = wmma32(load_af(ah_dt, 1), Bv[1], acc);
#pragma unroll
      for (int rr = 0; rr < 8; ++rr) {
        int row = rbase + rr;
        float v = acc[rr] > 0.f ? acc[rr] : 0.f;
        float dec = __expf(-v);
        sh_decay[row * 128 + colU] = dec;
        ah_decay[apos(row, colU)] = (_Float16)dec;
      }
    } else if (wqu < 12) {
#pragma unroll
      for (int e = 0; e < 8; ++e) zc[e] = b_cor_c;
      zc = wmma32(load_af(ah_xcur, 0), Bx[0], zc);
      zc = wmma32(load_af(ah_xcur, 1), Bx[1], zc);
#pragma unroll
      for (int rr = 0; rr < 8; ++rr)
        sh_zcorr[(rbase + rr) * 64 + colF] = zc[rr];
    }
    __syncthreads();

    // ---- stage 5: beta (w12-15); dh = decay*h -> f16 mirror (w0-7);
    //               deferred z_corr out stores (w8-11)
    if (wqu >= 12) {
      v8f acc;
#pragma unroll
      for (int e = 0; e < 8; ++e) acc[e] = b_cat0;
      acc = wmma32(load_af(ah_m, 0), Bx[0], acc);
      acc = wmma32(load_af(ah_m, 1), Bx[1], acc);
#pragma unroll
      for (int kc = 0; kc < 4; ++kc)
        acc = wmma32(load_af(ah_decay, kc), Bx[2 + kc], acc);
#pragma unroll
      for (int rr = 0; rr < 8; ++rr)
        sh_tvbeta[(rbase + rr) * 64 + colF] = fast_sigmoid(acc[rr]);
    } else if (wqu >= 8) {
#pragma unroll
      for (int rr = 0; rr < 8; ++rr)
        out[((size_t)t * B_ + (b0 + rbase + rr)) * X5F + 64 + colF] = zc[rr];
    } else {
#pragma unroll
      for (int i = 0; i < 8; ++i) {
        int idx = tid + i * 256;     // tid < 256 here
        int row = idx >> 7, u = idx & 127;
        ah_dh[apos(row, u)] = (_Float16)(sh_decay[idx] * sh_h[idx]);
      }
    }
    __syncthreads();

    // ---- stage 6: c = m*x_cur + (1-m)*(beta*z_corr + (1-beta)*x_reg)
    float o_c[2];
#pragma unroll
    for (int i = 0; i < 2; ++i) {
      int idx = tid + i * 512;
      int row = idx >> 6, f = idx & 63;
      int pf = apos(row, f);
      float beta = sh_tvbeta[idx];
      float ct = beta * sh_zcorr[idx] + (1.f - beta) * sh_xreg[idx];
      float mm = (float)ah_m[pf];
      float c = mm * sh_xcur[idx] + (1.f - mm) * ct;
      ah_cm[pf] = (_Float16)c;   // c-half of [c|m]
      o_c[i] = c;
    }
    __syncthreads();
    // deferred stage-6 output stores (overlap with stage-7 GEMMs)
#pragma unroll
    for (int i = 0; i < 2; ++i) {
      int idx = tid + i * 512;
      int row = idx >> 6, f = idx & 63;
      out[((size_t)t * B_ + (b0 + row)) * X5F + 128 + f] = o_c[i];
    }

    // ---- stage 7: z = sig(dh@Uu + cm@Wu + b) on w0-7;
    //               r = sig(dh@Ur + cm@Wr + b) on w8-15 (parallel)
    v8f zg;
    {
      v8f acc;
      const float bias7 = (wqu < 8) ? b_upd0 : b_rst0;
#pragma unroll
      for (int e = 0; e < 8; ++e) acc[e] = bias7;
#pragma unroll
      for (int kc = 0; kc < 4; ++kc) {
        acc = wmma32(load_af(ah_dh, kc), By[kc], acc);
        acc = wmma32(load_af(ah_cm, kc), Bz[kc], acc);
      }
      if (wqu < 8) {
#pragma unroll
        for (int rr = 0; rr < 8; ++rr) zg[rr] = fast_sigmoid(acc[rr]);
      } else {
#pragma unroll
        for (int rr = 0; rr < 8; ++rr)
          sh_decay[(rbase + rr) * 128 + colU2] = fast_sigmoid(acc[rr]);  // r
      }
    }
    __syncthreads();

    // ---- stage 8: rdh = r*dh (w0-7) ; Pvm = cm@Wvm + b_vm (w8-15, parallel)
    if (wqu >= 8) {
      v8f acc;
#pragma unroll
      for (int e = 0; e < 8; ++e) acc[e] = b_vm0;
#pragma unroll
      for (int kc = 0; kc < 4; ++kc) acc = wmma32(load_af(ah_cm, kc), Bv[kc], acc);
#pragma unroll
      for (int rr = 0; rr < 8; ++rr)
        sh_pvm[(rbase + rr) * 128 + colU2] = acc[rr];
    } else {
#pragma unroll
      for (int i = 0; i < 8; ++i) {
        int idx = tid + i * 256;     // tid < 256 here
        int row = idx >> 7, u = idx & 127;
        int p = apos(row, u);
        ah_dh[p] = (_Float16)((float)ah_dh[p] * sh_decay[idx]);
      }
    }
    __syncthreads();

    // ---- stage 9: h_hat = tanh(rdh@Uu + Pvm); h = (1-z)h + z*h_hat (w0-7)
    if (wqu < 8) {
      v8f acc;
#pragma unroll
      for (int rr = 0; rr < 8; ++rr)
        acc[rr] = sh_pvm[(rbase + rr) * 128 + colU];
#pragma unroll
      for (int kc = 0; kc < 4; ++kc) acc = wmma32(load_af(ah_dh, kc), By[kc], acc);
#pragma unroll
      for (int rr = 0; rr < 8; ++rr) {
        int row = rbase + rr;
        int idx = row * 128 + colU;
        float hh = fast_tanh(acc[rr]);
        float hn = (1.f - zg[rr]) * sh_h[idx] + zg[rr] * hh;
        sh_h[idx] = hn;
        ah_h[apos(row, colU)] = (_Float16)hn;
      }
    }
    __syncthreads();
  }

  // ---- final carries: hT, tsT, dtT appended after out_seq ----
  const size_t O1 = (size_t)T_ * B_ * X5F;
  for (int i = tid; i < 16 * 128; i += 512)
    out[O1 + (size_t)(b0 + (i >> 7)) * 128 + (i & 127)] = sh_h[i];
  for (int i = tid; i < 16 * 64; i += 512) {
    size_t g = (size_t)(b0 + (i >> 6)) * 64 + (i & 63);
    out[O1 + (size_t)B_ * 128 + g] = sh_ts[i];
    out[O1 + (size_t)B_ * 128 + (size_t)B_ * 64 + g] = sh_dt[i];
  }
}

extern "C" void kernel_launch(void* const* d_in, const int* in_sizes, int n_in,
                              void* d_out, int out_size, void* d_ws, size_t ws_size,
                              hipStream_t stream) {
  (void)in_sizes; (void)n_in; (void)out_size; (void)ws_size;
  const float* inputs       = (const float*)d_in[0];
  const float* h0           = (const float*)d_in[1];
  const float* ts0          = (const float*)d_in[2];
  const float* dt0          = (const float*)d_in[3];
  const float* w_regression = (const float*)d_in[4];
  const float* b_regression = (const float*)d_in[5];
  const float* w_controleur = (const float*)d_in[6];
  const float* b_controleur = (const float*)d_in[7];
  const float* w_delta      = (const float*)d_in[8];
  const float* b_delta      = (const float*)d_in[9];
  const float* w_value_mask = (const float*)d_in[10];
  const float* b_value_mask = (const float*)d_in[11];
  const float* w_corr       = (const float*)d_in[12];
  const float* b_corr       = (const float*)d_in[13];
  const float* w_concat     = (const float*)d_in[14];
  const float* b_concat     = (const float*)d_in[15];
  const float* u_update     = (const float*)d_in[16];
  const float* w_update     = (const float*)d_in[17];
  const float* b_update     = (const float*)d_in[18];
  const float* u_reset      = (const float*)d_in[19];
  const float* w_reset      = (const float*)d_in[20];
  const float* b_reset      = (const float*)d_in[21];
  float* out = (float*)d_out;
  _Float16* ws = (_Float16*)d_ws;

  auto packLaunch = [&](const float* W, int ofs, int K, int N, int mask) {
    int total = K * N;
    pack_weights<<<(total + 255) / 256, 256, 0, stream>>>(W, ws + ofs, K, N, mask);
  };
  packLaunch(w_regression, OFS_WREG, 128, 64, 0);
  packLaunch(w_controleur, OFS_WCTL, 128, 64, 0);
  packLaunch(w_delta,      OFS_WDEL, 64, 128, 0);
  packLaunch(w_value_mask, OFS_WVM,  128, 128, 0);
  packLaunch(w_corr,       OFS_WCORR, 64, 64, 1);
  packLaunch(w_concat,     OFS_WCAT, 192, 64, 0);
  packLaunch(u_update,     OFS_UU,   128, 128, 0);
  packLaunch(w_update,     OFS_WU,   128, 128, 0);
  packLaunch(u_reset,      OFS_UR,   128, 128, 0);
  packLaunch(w_reset,      OFS_WR,   128, 128, 0);

  padd_gru<<<16, 512, 0, stream>>>(inputs, h0, ts0, dt0,
                                   b_regression, b_controleur, b_delta, b_value_mask,
                                   b_corr, b_concat, b_update, b_reset, ws, out);
}
